// AR_30056181137502
// MI455X (gfx1250) — matrix operand order repaired and checked by reading, not compiled, not run on hardware
//
#include <hip/hip_runtime.h>
#include <hip/hip_bf16.h>

typedef _Float16 f16;
typedef _Float16 v8h  __attribute__((ext_vector_type(8)));
typedef _Float16 v16h __attribute__((ext_vector_type(16)));
typedef float    v8f  __attribute__((ext_vector_type(8)));

constexpr int kL = 6;
constexpr int kD = 768;
constexpr int kH = 12;
constexpr int kHD = 64;
constexpr int kFF = 3072;
constexpr int kV = 2051;
constexpr int kP = 128;        // prompt length
constexpr int kMaxLen = 16;
constexpr int kT = kP + kMaxLen;   // 144 live rows
constexpr int kTPad = 160;         // padded rows (5 x 32-row GEMM blocks)

// ---------------------------------------------------------------------------
// Weight conversion: f32 [K][N] -> f16 [N][K] (B operand, N-major, K contiguous)
// ---------------------------------------------------------------------------
__global__ void conv_tr_kernel(const float* __restrict__ in, f16* __restrict__ out,
                               int N, int K) {
  size_t idx = (size_t)blockIdx.x * 256 + threadIdx.x;
  if (idx >= (size_t)N * K) return;
  int n = (int)(idx / K), k = (int)(idx % K);
  out[idx] = (f16)in[(size_t)k * N + n];
}

__global__ void conv_kernel(const float* __restrict__ in, f16* __restrict__ out, int n) {
  int i = blockIdx.x * 256 + threadIdx.x;
  if (i < n) out[i] = (f16)in[i];
}

// ---------------------------------------------------------------------------
// Token buffer init (prompt copy + zero decode slots) — once per launch
// ---------------------------------------------------------------------------
__global__ void init_kernel(int* __restrict__ idx, float* __restrict__ out,
                            const int* __restrict__ first_idx) {
  int i = threadIdx.x;
  if (i < kP) { idx[i] = first_idx[i]; out[i] = (float)first_idx[i]; }
  else if (i < kTPad) { idx[i] = 0; if (i < kT) out[i] = 0.f; }
}

// ---------------------------------------------------------------------------
// Embedding: h[t] = wte[tok[t]] + wpe[pos[t]] ; zero the padded rows
// ---------------------------------------------------------------------------
__global__ __launch_bounds__(256)
void embed_kernel(float* __restrict__ h, const float* __restrict__ wte,
                  const float* __restrict__ wpe, const int* __restrict__ idx,
                  const int* __restrict__ pos_ids) {
  int t = blockIdx.x;
  int tok = 0, p = 0;
  if (t < kT) {
    tok = idx[t];
    p = (t < kP) ? pos_ids[t] : pos_ids[kP - 1] + 1 + (t - kP);
  }
#pragma unroll
  for (int i = 0; i < 3; ++i) {
    int d = threadIdx.x + i * 256;
    float v = 0.f;
    if (t < kT) v = wte[(size_t)tok * kD + d] + wpe[(size_t)p * kD + d];
    h[(size_t)t * kD + d] = v;
  }
}

// ---------------------------------------------------------------------------
// LayerNorm (per row), f32 in -> f16 out (GEMM A operand)
// ---------------------------------------------------------------------------
__global__ __launch_bounds__(256)
void ln_kernel(const float* __restrict__ h, const float* __restrict__ g,
               const float* __restrict__ b, f16* __restrict__ out) {
  __shared__ float s1[256], s2[256];
  int t = blockIdx.x;
  const float* row = h + (size_t)t * kD;
  float x[3], sum = 0.f, sq = 0.f;
#pragma unroll
  for (int i = 0; i < 3; ++i) {
    x[i] = row[threadIdx.x + i * 256];
    sum += x[i]; sq += x[i] * x[i];
  }
  s1[threadIdx.x] = sum; s2[threadIdx.x] = sq;
  __syncthreads();
  for (int off = 128; off > 0; off >>= 1) {
    if (threadIdx.x < off) {
      s1[threadIdx.x] += s1[threadIdx.x + off];
      s2[threadIdx.x] += s2[threadIdx.x + off];
    }
    __syncthreads();
  }
  float mean = s1[0] * (1.f / kD);
  float var  = s2[0] * (1.f / kD) - mean * mean;
  float rstd = rsqrtf(var + 1e-5f);
#pragma unroll
  for (int i = 0; i < 3; ++i) {
    int d = threadIdx.x + i * 256;
    out[(size_t)t * kD + d] = (f16)((x[i] - mean) * rstd * g[d] + b[d]);
  }
}

// ---------------------------------------------------------------------------
// WMMA GEMM: C[160][N] = A_f16[160][K] @ Bt_f16[N][K]^T + bias
// EPI 0: store f32   EPI 1: residual add into f32   EPI 2: gelu -> f16
// Block = 128 threads = 4 waves; each wave owns a 32(M) x 16(N) slab as two
// 16x16 WMMA tiles sharing one B fetch per K-chunk (halves weight traffic).
// A/B lane layouts follow the CDNA5 16-bit WMMA operand maps, so operands are
// fetched with plain 16-byte global loads (global_load_b128), no shuffling.
// The weight stream is prefetched one chunk-group ahead (global_prefetch_b8);
// speculative prefetch past the row end is dropped silently per the ISA.
// ---------------------------------------------------------------------------
__device__ __forceinline__ float gelu_tanh(float v) {
  float x3 = v * v * v;
  return 0.5f * v * (1.f + tanhf(0.7978845608028654f * (v + 0.044715f * x3)));
}

template <int EPI>
__global__ __launch_bounds__(128)
void gemm_wmma(const f16* __restrict__ A, const f16* __restrict__ Bt,
               const float* __restrict__ bias, f16* __restrict__ outH,
               float* __restrict__ outF, int N, int K) {
  const int lane = threadIdx.x & 31;
  const int w    = threadIdx.x >> 5;          // 0..3 -> N-tile within block
  const int l15  = lane & 15;
  const int hi   = lane >> 4;                 // 0: lanes 0-15, 1: lanes 16-31
  const int tileM = blockIdx.y * 32;
  const int tileN = blockIdx.x * 64 + w * 16;

  // A: row M = tile + l15; halves [0..7] = K kc+8*hi.., halves [8..15] = +16
  const f16* pA0 = A + (size_t)(tileM + l15) * K + hi * 8;
  const f16* pA1 = A + (size_t)(tileM + 16 + l15) * K + hi * 8;
  // B: col N = tileN + l15; halves [0..7] = K kc+8*hi.., halves [8..15] = +16
  const f16* pB  = Bt + (size_t)(tileN + l15) * K + hi * 8;

  v8f c0 = {}, c1 = {};
  for (int kc = 0; kc < K; kc += 32) {
    __builtin_prefetch(pB + kc + 256, 0, 0);   // weight stream, 512B ahead
    v8h b0 = *(const v8h*)(pB + kc);
    v8h b1 = *(const v8h*)(pB + kc + 16);
    v8h a00 = *(const v8h*)(pA0 + kc);
    v8h a01 = *(const v8h*)(pA0 + kc + 16);
    v8h a10 = *(const v8h*)(pA1 + kc);
    v8h a11 = *(const v8h*)(pA1 + kc + 16);
    v16h b, a0, a1;
#pragma unroll
    for (int i = 0; i < 8; ++i) {
      b[i]  = b0[i];  b[i + 8]  = b1[i];
      a0[i] = a00[i]; a0[i + 8] = a01[i];
      a1[i] = a10[i]; a1[i + 8] = a11[i];
    }
    c0 = __builtin_amdgcn_wmma_f32_16x16x32_f16(false, a0, false, b,
                                                (short)0, c0, false, false);
    c1 = __builtin_amdgcn_wmma_f32_16x16x32_f16(false, a1, false, b,
                                                (short)0, c1, false, false);
  }

  const int col = tileN + l15;
  const float bval = bias[col];
#pragma unroll
  for (int r = 0; r < 8; ++r) {
    int row0 = tileM + hi * 8 + r;       // C layout: VGPR r -> M = r (+8 hi lanes)
    int row1 = row0 + 16;
    float v0 = c0[r] + bval;
    float v1 = c1[r] + bval;
    if (EPI == 0) {
      outF[(size_t)row0 * N + col] = v0;
      outF[(size_t)row1 * N + col] = v1;
    } else if (EPI == 1) {
      outF[(size_t)row0 * N + col] += v0;
      outF[(size_t)row1 * N + col] += v1;
    } else {
      outH[(size_t)row0 * N + col] = (f16)gelu_tanh(v0);
      outH[(size_t)row1 * N + col] = (f16)gelu_tanh(v1);
    }
  }
}

// ---------------------------------------------------------------------------
// Attention: one block per (head, query row). qkv f32 [160][3D] -> o f16 [160][D]
// ---------------------------------------------------------------------------
__global__ __launch_bounds__(128)
void attn_kernel(const float* __restrict__ qkv, f16* __restrict__ o) {
  const int hh = blockIdx.x, t = blockIdx.y, tid = threadIdx.x;
  if (t >= kT) {            // keep padded rows defined (uniform branch, no syncs)
    if (tid < kHD) o[(size_t)t * kD + hh * kHD + tid] = (f16)0.f;
    return;
  }
  __shared__ float qs[kHD];
  __shared__ float s[kT];
  __shared__ float red[128];
  if (tid < kHD) qs[tid] = qkv[(size_t)t * (3 * kD) + hh * kHD + tid];
  __syncthreads();
  for (int j = tid; j <= t; j += 128) {
    const float* kr = qkv + (size_t)j * (3 * kD) + kD + hh * kHD;
    float dot = 0.f;
#pragma unroll
    for (int d = 0; d < kHD; ++d) dot += qs[d] * kr[d];
    s[j] = dot * 0.125f;   // 1/sqrt(64)
  }
  __syncthreads();
  float pm = -INFINITY;
  for (int j = tid; j <= t; j += 128) pm = fmaxf(pm, s[j]);
  red[tid] = pm; __syncthreads();
  for (int off = 64; off > 0; off >>= 1) {
    if (tid < off) red[tid] = fmaxf(red[tid], red[tid + off]);
    __syncthreads();
  }
  float m = red[0]; __syncthreads();
  float ps = 0.f;
  for (int j = tid; j <= t; j += 128) { float e = __expf(s[j] - m); s[j] = e; ps += e; }
  red[tid] = ps; __syncthreads();
  for (int off = 64; off > 0; off >>= 1) {
    if (tid < off) red[tid] += red[tid + off];
    __syncthreads();
  }
  float inv = 1.f / red[0];
  if (tid < kHD) {
    float acc = 0.f;
    for (int j = 0; j <= t; ++j)
      acc += s[j] * qkv[(size_t)j * (3 * kD) + 2 * kD + hh * kHD + tid];
    o[(size_t)t * kD + hh * kHD + tid] = (f16)(acc * inv);
  }
}

// ---------------------------------------------------------------------------
// LM head GEMV for one row: logits[n] = x(cur-1) . wte16[n]
// ---------------------------------------------------------------------------
__global__ __launch_bounds__(256)
void lmhead_kernel(const f16* __restrict__ xrow, const f16* __restrict__ wte16,
                   float* __restrict__ logits) {
  __shared__ f16 xs[kD];
#pragma unroll
  for (int i = 0; i < 3; ++i) xs[threadIdx.x + i * 256] = xrow[threadIdx.x + i * 256];
  __syncthreads();
  int n = blockIdx.x * 256 + threadIdx.x;
  if (n >= kV) return;
  const f16* wr = wte16 + (size_t)n * kD;
  float acc = 0.f;
  for (int d = 0; d < kD; d += 8) {
    v8h wv = *(const v8h*)(wr + d);
#pragma unroll
    for (int i = 0; i < 8; ++i) acc += (float)wv[i] * (float)xs[d + i];
  }
  logits[n] = acc;
}

// ---------------------------------------------------------------------------
// Mask + top-k threshold + gumbel-argmax sample + emit step logits & token.
// ---------------------------------------------------------------------------
__global__ __launch_bounds__(1024)
void finalize_kernel(const float* __restrict__ logits_in,
                     const int* __restrict__ topk_p, int* __restrict__ idx,
                     float* __restrict__ out, int step, int cur) {
  __shared__ float L[kV];
  __shared__ float Wk[kV];
  __shared__ float rv[1024];
  __shared__ int   ri[1024];
  __shared__ float kthv;
  const int tid = threadIdx.x;
  for (int i = tid; i < kV; i += 1024) {
    float v = logits_in[i];                           // TEMP == 1.0
    bool masked = (i <= 1024) || (i == 1024 + 1 + 1024);
    v = masked ? -INFINITY : v;
    L[i] = v; Wk[i] = v;
  }
  __syncthreads();
  int kk = topk_p[0]; if (kk > kV) kk = kV;
  for (int it = 0; it < kk; ++it) {
    float bv = -INFINITY; int bi = -1;
    for (int i = tid; i < kV; i += 1024)
      if (Wk[i] > bv) { bv = Wk[i]; bi = i; }
    rv[tid] = bv; ri[tid] = bi;
    __syncthreads();
    for (int off = 512; off > 0; off >>= 1) {
      if (tid < off && rv[tid + off] > rv[tid]) { rv[tid] = rv[tid + off]; ri[tid] = ri[tid + off]; }
      __syncthreads();
    }
    if (tid == 0) { kthv = rv[0]; if (ri[0] >= 0) Wk[ri[0]] = -INFINITY; }
    __syncthreads();
  }
  float kth = kthv;
  for (int i = tid; i < kV; i += 1024) {
    float v = (L[i] < kth) ? -INFINITY : L[i];
    out[kT + (size_t)step * kV + i] = v;
    L[i] = v;
  }
  __syncthreads();
  // deterministic gumbel-argmax over surviving logits
  float bv = -INFINITY; int bi = 0;
  for (int i = tid; i < kV; i += 1024) {
    float lv = L[i];
    if (lv == -INFINITY) continue;
    unsigned h = (unsigned)(i * 2654435761u) ^ (unsigned)(step * 2246822519u) ^ 0x9E3779B9u;
    h ^= h >> 16; h *= 0x7feb352du; h ^= h >> 15; h *= 0x846ca68bu; h ^= h >> 16;
    float u = (float)(h >> 8) * (1.0f / 16777216.0f) + 1e-7f;
    float sc = lv - __logf(-__logf(u));
    if (sc > bv) { bv = sc; bi = i; }
  }
  rv[tid] = bv; ri[tid] = bi;
  __syncthreads();
  for (int off = 512; off > 0; off >>= 1) {
    if (tid < off && rv[tid + off] > rv[tid]) { rv[tid] = rv[tid + off]; ri[tid] = ri[tid + off]; }
    __syncthreads();
  }
  if (tid == 0) { idx[cur] = ri[0]; out[cur] = (float)ri[0]; }
}

// ---------------------------------------------------------------------------
// Launch: convert weights once, then 16 full-sequence decode steps.
// ---------------------------------------------------------------------------
extern "C" void kernel_launch(void* const* d_in, const int* in_sizes, int n_in,
                              void* d_out, int out_size, void* d_ws, size_t ws_size,
                              hipStream_t stream) {
  (void)in_sizes; (void)n_in; (void)out_size; (void)ws_size;
  const float* wte   = (const float*)d_in[0];
  const float* wpe   = (const float*)d_in[1];
  const float* ln1_g = (const float*)d_in[2];
  const float* ln1_b = (const float*)d_in[3];
  const float* attn_w = (const float*)d_in[4];
  const float* attn_b = (const float*)d_in[5];
  const float* proj_w = (const float*)d_in[6];
  const float* proj_b = (const float*)d_in[7];
  const float* ln2_g = (const float*)d_in[8];
  const float* ln2_b = (const float*)d_in[9];
  const float* fc_w  = (const float*)d_in[10];
  const float* fc_b  = (const float*)d_in[11];
  const float* fcp_w = (const float*)d_in[12];
  const float* fcp_b = (const float*)d_in[13];
  const float* lnf_g = (const float*)d_in[14];
  const float* lnf_b = (const float*)d_in[15];
  const int* first_idx = (const int*)d_in[16];
  const int* pos_ids   = (const int*)d_in[17];
  const int* topk_p    = (const int*)d_in[19];
  float* out = (float*)d_out;

  char* ws = (char*)d_ws;
  size_t off = 0;
  auto alloc = [&](size_t bytes) -> void* {
    void* p = ws + off;
    off = (off + bytes + 255) & ~(size_t)255;
    return p;
  };
  f16* wqkvT = (f16*)alloc(sizeof(f16) * (size_t)kL * 3 * kD * kD);
  f16* wprojT = (f16*)alloc(sizeof(f16) * (size_t)kL * kD * kD);
  f16* wfcT  = (f16*)alloc(sizeof(f16) * (size_t)kL * kFF * kD);
  f16* wfcpT = (f16*)alloc(sizeof(f16) * (size_t)kL * kD * kFF);
  f16* wte16 = (f16*)alloc(sizeof(f16) * (size_t)kV * kD);
  f16* xbuf  = (f16*)alloc(sizeof(f16) * (size_t)kTPad * kD);
  f16* obuf  = (f16*)alloc(sizeof(f16) * (size_t)kTPad * kD);
  f16* ffbuf = (f16*)alloc(sizeof(f16) * (size_t)kTPad * kFF);
  f16* xfbuf = (f16*)alloc(sizeof(f16) * (size_t)kTPad * kD);
  float* hbuf   = (float*)alloc(sizeof(float) * (size_t)kTPad * kD);
  float* qkvbuf = (float*)alloc(sizeof(float) * (size_t)kTPad * 3 * kD);
  float* logits = (float*)alloc(sizeof(float) * 2052);
  int* idxbuf   = (int*)alloc(sizeof(int) * kTPad);

  // one-time (per launch) weight convert + transpose to f16 [N][K]
  for (int l = 0; l < kL; ++l) {
    size_t c0 = (size_t)3 * kD * kD;
    conv_tr_kernel<<<(c0 + 255) / 256, 256, 0, stream>>>(
        attn_w + (size_t)l * 3 * kD * kD, wqkvT + (size_t)l * 3 * kD * kD, 3 * kD, kD);
    size_t c1 = (size_t)kD * kD;
    conv_tr_kernel<<<(c1 + 255) / 256, 256, 0, stream>>>(
        proj_w + (size_t)l * kD * kD, wprojT + (size_t)l * kD * kD, kD, kD);
    size_t c2 = (size_t)kFF * kD;
    conv_tr_kernel<<<(c2 + 255) / 256, 256, 0, stream>>>(
        fc_w + (size_t)l * kD * kFF, wfcT + (size_t)l * kFF * kD, kFF, kD);
    size_t c3 = (size_t)kD * kFF;
    conv_tr_kernel<<<(c3 + 255) / 256, 256, 0, stream>>>(
        fcp_w + (size_t)l * kFF * kD, wfcpT + (size_t)l * kD * kFF, kD, kFF);
  }
  conv_kernel<<<((size_t)kV * kD + 255) / 256, 256, 0, stream>>>(wte, wte16, kV * kD);
  init_kernel<<<1, 256, 0, stream>>>(idxbuf, out, first_idx);

  const dim3 gQKV(3 * kD / 64, kTPad / 32);
  const dim3 gD(kD / 64, kTPad / 32);
  const dim3 gFF(kFF / 64, kTPad / 32);

  for (int s = 0; s < kMaxLen; ++s) {
    const int cur = kP + s;
    embed_kernel<<<kTPad, 256, 0, stream>>>(hbuf, wte, wpe, idxbuf, pos_ids);
    for (int l = 0; l < kL; ++l) {
      ln_kernel<<<kTPad, 256, 0, stream>>>(hbuf, ln1_g + l * kD, ln1_b + l * kD, xbuf);
      gemm_wmma<0><<<gQKV, 128, 0, stream>>>(xbuf, wqkvT + (size_t)l * 3 * kD * kD,
                                             attn_b + l * 3 * kD, nullptr, qkvbuf,
                                             3 * kD, kD);
      attn_kernel<<<dim3(kH, kTPad), 128, 0, stream>>>(qkvbuf, obuf);
      gemm_wmma<1><<<gD, 128, 0, stream>>>(obuf, wprojT + (size_t)l * kD * kD,
                                           proj_b + l * kD, nullptr, hbuf, kD, kD);
      ln_kernel<<<kTPad, 256, 0, stream>>>(hbuf, ln2_g + l * kD, ln2_b + l * kD, xbuf);
      gemm_wmma<2><<<gFF, 128, 0, stream>>>(xbuf, wfcT + (size_t)l * kFF * kD,
                                            fc_b + l * kFF, ffbuf, nullptr, kFF, kD);
      gemm_wmma<1><<<gD, 128, 0, stream>>>(ffbuf, wfcpT + (size_t)l * kD * kFF,
                                           fcp_b + l * kD, nullptr, hbuf, kD, kFF);
    }
    ln_kernel<<<kTPad, 256, 0, stream>>>(hbuf, lnf_g, lnf_b, xfbuf);
    lmhead_kernel<<<(kV + 255) / 256, 256, 0, stream>>>(
        xfbuf + (size_t)(cur - 1) * kD, wte16, logits);
    finalize_kernel<<<1, 1024, 0, stream>>>(logits, topk_p, idxbuf, out, s, cur);
  }
}
